// Hosvd_67568425501233
// MI455X (gfx1250) — compile-verified, matrix operand order checked by reading, not run on hardware
//
#include <hip/hip_runtime.h>

typedef __attribute__((ext_vector_type(16))) _Float16 v16h;
typedef __attribute__((ext_vector_type(8)))  _Float16 v8h;
typedef __attribute__((ext_vector_type(8)))  float    v8f;
typedef __attribute__((ext_vector_type(4)))  unsigned int v4u;
typedef __attribute__((ext_vector_type(4)))  int v4i;
typedef __attribute__((ext_vector_type(8)))  int v8i;

#define HS   256   // h_size
#define NCH  2     // max_output_degree
#define TSZ  32    // t_size
#define NA   4     // n_aggr
#define RK   16    // rank
#define TRK  8     // t_rank
#define BS   16384
#define KBIG 153   // 9*17  (p,q) flattened
#define KPAD 160   // padded to multiple of 32
#define NBIG 272   // 17*16 (r,s) flattened

#define T_TILES  85   // (j=17) x (kt=5), tile = j*5 + kt
#define U_TILES  16   // (c=2) x (kt=8), tile = c*8 + kt
#define UO_TILES 16   // nt = 0..15
#define TILE_H   512  // halfs per swizzled B tile (32 lanes x 16 halfs)
#define WS_PER_A ((T_TILES + U_TILES + UO_TILES) * TILE_H)   // 59904 halfs per aggregator

#define WMMA_F32_F16(a, b, c) \
  __builtin_amdgcn_wmma_f32_16x16x32_f16(false, (a), false, (b), (short)0, (c), false, false)

// ---------------------------------------------------------------------------
// Swizzled B-tile layout: per tile, plane0 = element 0..7 of each lane
// (lane-contiguous 16B), plane1 = element 8..15.  One fragment = 2x b128.
//   khi = (lane>>4)*8 ;  ep<8 -> k=khi+ep ;  ep>=8 -> k=8+khi+ep
// ---------------------------------------------------------------------------
__device__ __forceinline__ int bswz_idx(int tile, int klocal, int col) {
  const int lane = col | (((klocal >> 3) & 1) << 4);
  const int ep   = (klocal & 7) | ((klocal >> 4) << 3);
  return tile * TILE_H + ((ep >> 3) << 8) + lane * 8 + (ep & 7);
}

__device__ __forceinline__ v16h load_bfrag(const _Float16* tb, int lane) {
  const v8h lo = *(const v8h*)(tb + lane * 8);         // ds_load_b128
  const v8h hi = *(const v8h*)(tb + 256 + lane * 8);   // ds_load_b128
  v16h f;
#pragma unroll
  for (int i = 0; i < 8; ++i) { f[i] = lo[i]; f[8 + i] = hi[i]; }
  return f;
}

// A fragment (16x32 f16) from row-major f32 global memory: 4x b128 + cvt_pk.
__device__ __forceinline__ v16h frag_A_gf32(const float* base, int ld, int lane) {
  const int row = lane & 15;
  const int khi = (lane >> 4) << 3;
  const float* p = base + (size_t)row * ld + khi;
  float t[16];
  *(float4*)(t + 0)  = *(const float4*)(p + 0);
  *(float4*)(t + 4)  = *(const float4*)(p + 4);
  *(float4*)(t + 8)  = *(const float4*)(p + 16);
  *(float4*)(t + 12) = *(const float4*)(p + 20);
  v16h f;
#pragma unroll
  for (int i = 0; i < 16; ++i) f[i] = (_Float16)t[i];
  return f;
}

// Build the 5 Tucker A fragments (w = h0 (x) h1) entirely in registers.
template <int KHI>
__device__ __forceinline__ void build_w_frags(const float* h0r, const float* h1r, v16h* af) {
#pragma unroll
  for (int kt = 0; kt < 5; ++kt) {
    v16h f;
#pragma unroll
    for (int e = 0; e < 16; ++e) {
      const int k = kt * 32 + KHI + e + (e >= 8 ? 8 : 0);
      float v = 0.0f;
      if (k < KBIG) {
        const int p = k / 17, q = k - p * 17;
        v = h0r[p] * h1r[q];
      }
      f[e] = (_Float16)v;
    }
    af[kt] = f;
  }
}

// ---------------------------------------------------------------------------
// Prep kernel (once per launch, 4 blocks): build the pre-swizzled f16
// B-fragment image per aggregator in workspace.  Hoists all swizzle index
// math + f32->f16 conversion out of the 1024 compute blocks.
// ---------------------------------------------------------------------------
__global__ void hosvd_prep_ws(const float* __restrict__ U,
                              const float* __restrict__ T,
                              const float* __restrict__ Uo,
                              _Float16* __restrict__ ws) {
  const int a = blockIdx.x;
  _Float16* Tw  = ws + (size_t)a * WS_PER_A;
  _Float16* Uw  = Tw + T_TILES * TILE_H;
  _Float16* Uow = Uw + U_TILES * TILE_H;
  // T[a] reshaped (153 x 272), K-padded to 160 rows of zeros.
  for (int idx = threadIdx.x; idx < KPAD * NBIG; idx += blockDim.x) {
    const int k = idx / NBIG, n = idx - k * NBIG;
    const float v = (k < KBIG) ? T[(size_t)a * (KBIG * NBIG) + k * NBIG + n] : 0.0f;
    Tw[bswz_idx((n >> 4) * 5 + (k >> 5), k & 31, n & 15)] = (_Float16)v;
  }
  // U[c][a] (256 x 16)
  for (int idx = threadIdx.x; idx < 2 * 256 * 16; idx += blockDim.x) {
    const int c = idx >> 12, rem = idx & 4095;
    const int k = rem >> 4;
    Uw[bswz_idx(c * 8 + (k >> 5), k & 31, rem & 15)] =
        (_Float16)U[((size_t)(c * NA + a) << 12) + rem];
  }
  // U_output[a] (16 x 256), K padded to 32 rows of zeros
  for (int idx = threadIdx.x; idx < 32 * 256; idx += blockDim.x) {
    const int s = idx >> 8, h = idx & 255;
    const float v = (s < 16) ? Uo[(size_t)(a * 16 + s) * 256 + h] : 0.0f;
    Uow[bswz_idx(h >> 4, s, h & 15)] = (_Float16)v;
  }
}

__global__ void hosvd_tucker_wmma(const float* __restrict__ nh,   // (BS,2,256)
                                  const float* __restrict__ te,   // (BS,32)
                                  const float* __restrict__ bU,   // (2,4,1,16)
                                  const float* __restrict__ Ut,   // (32,32)
                                  const float* __restrict__ bt,   // (1,32)
                                  const float* __restrict__ bo,   // (4,1,256)
                                  const _Float16* __restrict__ ws,
                                  float* __restrict__ out) {      // (BS,1024)
  extern __shared__ char smem_raw[];
  _Float16* Tl  = (_Float16*)smem_raw;         // contiguous: must match ws order
  _Float16* Ul  = Tl + T_TILES * TILE_H;
  _Float16* Uol = Ul + U_TILES * TILE_H;
  float* h0l = (float*)(Uol + UO_TILES * TILE_H); // [4 waves][16][12]
  float* h1l = h0l + 4 * 16 * 12;                 // [4 waves][16][20]
  float* h2t = h1l + 4 * 16 * 20;                 // [4 waves][17][16]  ([r][row])
  float* gl  = h2t + 4 * 17 * 16;                 // [4 waves][16][32]  f32

  const int tid   = threadIdx.x;
  const int lane  = tid & 31;
  const int wv    = tid >> 5;
  const int a     = blockIdx.y;
  const int row0  = blockIdx.x * 64 + wv * 16;
  const int col   = lane & 15;
  const int rbase = (lane >> 4) << 3;

  // ---- staging: one contiguous 117KB ws->LDS fill via the Tensor Data Mover ----
  const _Float16* wsa = ws + (size_t)a * WS_PER_A;
#if __has_builtin(__builtin_amdgcn_tensor_load_to_lds) && __has_builtin(__builtin_amdgcn_s_wait_tensorcnt)
  if (wv == 0) {
    const unsigned long long ga = (unsigned long long)(uintptr_t)wsa;
    const unsigned ldsa = (unsigned)(uintptr_t)Tl;
    const unsigned n = WS_PER_A;                       // 59904 2-byte elements
    // D# group0: count=1 | lds_addr | global_addr[56:0] | type=2
    v4u g0 = { 1u, ldsa, (unsigned)ga,
               (unsigned)((ga >> 32) & 0x01FFFFFFu) | 0x80000000u };
    // D# group1: data_size=1(2B); tensor_dim0=n; tensor_dim1=1; tile_dim0=n;
    //            tile_dim1=1; tensor_dim0_stride=n
    v8i g1 = { (int)(1u << 16),
               (int)((n & 0xFFFFu) << 16),
               (int)(((n >> 16) & 0xFFFFu) | (1u << 16)),
               (int)((n & 0xFFFFu) << 16),
               1, (int)n, 0, 0 };
    v4i z4 = { 0, 0, 0, 0 };
    v8i z8 = { 0, 0, 0, 0, 0, 0, 0, 0 };
    // 6-arg form on this toolchain: (g0, g1, g2, g3, g4, cpol)
    __builtin_amdgcn_tensor_load_to_lds(g0, g1, z4, z4, z8, 0);
    __builtin_amdgcn_s_wait_tensorcnt(0);              // TENSORcnt == 0
  }
  __syncthreads();
#else
  for (int idx = tid; idx < WS_PER_A / 8; idx += blockDim.x)
    *(v8h*)(Tl + idx * 8) = *(const v8h*)(wsa + idx * 8);
  __syncthreads();
#endif

  // ---- stage 1: child projections -> h1 (row-major) and h2 (transposed) ----
#pragma unroll
  for (int c = 0; c < 2; ++c) {
    v8f acc = {};
#pragma unroll
    for (int kt = 0; kt < 8; ++kt) {
      v16h af = frag_A_gf32(nh + (size_t)row0 * (NCH * HS) + c * HS + kt * 32, NCH * HS, lane);
      v16h bf = load_bfrag(Ul + (c * 8 + kt) * TILE_H, lane);
      acc = WMMA_F32_F16(af, bf, acc);
    }
    const float bias = bU[(c * NA + a) * RK + col];
    if (c == 0) {
      float* h1w = h1l + wv * 320;
#pragma unroll
      for (int v = 0; v < 8; ++v) h1w[(v + rbase) * 20 + col] = acc[v] + bias;
      if (lane < 16) h1w[lane * 20 + 16] = 1.0f;           // homogeneous coord
    } else {
      float* h2w = h2t + wv * 272;
#pragma unroll
      for (int v = 0; v < 8; ++v) h2w[col * 16 + rbase + v] = acc[v] + bias;
      if (lane < 16) h2w[16 * 16 + lane] = 1.0f;           // homogeneous coord
    }
  }

  // ---- stage 2: type projection -> h0 (augmented); K=32 too small for WMMA ----
  if (lane < 16) {
    const int r = lane;
    const float* terow = te + (size_t)(row0 + r) * TSZ;
    float* h0w = h0l + (wv * 16 + r) * 12;
#pragma unroll
    for (int j = 0; j < TRK; ++j) {
      float acc = bt[a * TRK + j];
#pragma unroll
      for (int t = 0; t < TSZ; ++t) acc += terow[t] * Ut[t * (NA * TRK) + a * TRK + j];
      h0w[j] = acc;
    }
    h0w[8] = 1.0f;
  }
  __syncthreads();

  // ---- stage 3: Tucker A fragments (w = h0 (x) h1) built in registers ----
  float h0r[12], h1r[20];
  {
    const float* h0p = h0l + (wv * 16 + col) * 12;
    *(float4*)(h0r + 0) = *(const float4*)(h0p + 0);
    *(float4*)(h0r + 4) = *(const float4*)(h0p + 4);
    *(float4*)(h0r + 8) = *(const float4*)(h0p + 8);
    const float* h1p = h1l + wv * 320 + col * 20;
#pragma unroll
    for (int i = 0; i < 5; ++i) *(float4*)(h1r + 4 * i) = *(const float4*)(h1p + 4 * i);
  }
  v16h af[5];
  if (lane < 16) build_w_frags<0>(h0r, h1r, af);
  else           build_w_frags<8>(h0r, h1r, af);

  // ---- stage 4: (16x160)x(160x272) WMMA GEMM with fused h2 contraction ----
  v8f gacc = {};
  const float* h2w = h2t + wv * 272;
  for (int j = 0; j < 17; ++j) {                 // N-tile j == r index
    const _Float16* tb = Tl + (j * 5) * TILE_H;
    v8f acc = {};
    acc = WMMA_F32_F16(af[0], load_bfrag(tb + 0 * TILE_H, lane), acc);
    acc = WMMA_F32_F16(af[1], load_bfrag(tb + 1 * TILE_H, lane), acc);
    acc = WMMA_F32_F16(af[2], load_bfrag(tb + 2 * TILE_H, lane), acc);
    acc = WMMA_F32_F16(af[3], load_bfrag(tb + 3 * TILE_H, lane), acc);
    acc = WMMA_F32_F16(af[4], load_bfrag(tb + 4 * TILE_H, lane), acc);
    float hv[8];
    const float* h2p = h2w + j * 16 + rbase;
    *(float4*)(hv + 0) = *(const float4*)(h2p + 0);
    *(float4*)(hv + 4) = *(const float4*)(h2p + 4);
#pragma unroll
    for (int v = 0; v < 8; ++v) gacc[v] += hv[v] * acc[v];
  }

  // ---- stage 5: output projection g(16x16, K->32 zero pad) x Uo(32x256) ----
  float* glw = gl + wv * 512;
#pragma unroll
  for (int v = 0; v < 8; ++v) glw[(v + rbase) * 32 + col] = gacc[v];
  __syncthreads();
  v16h ga;
  {
    const float* gp = glw + col * 32 + rbase;
    float t[8];
    *(float4*)(t + 0) = *(const float4*)(gp + 0);
    *(float4*)(t + 4) = *(const float4*)(gp + 4);
#pragma unroll
    for (int i = 0; i < 8; ++i) { ga[i] = (_Float16)t[i]; ga[8 + i] = (_Float16)0.0f; }
  }
  for (int nt = 0; nt < 16; ++nt) {
    v16h bf = load_bfrag(Uol + nt * TILE_H, lane);
    v8f acc = {};
    acc = WMMA_F32_F16(ga, bf, acc);
    const float bias = bo[a * HS + nt * 16 + col];
#pragma unroll
    for (int v = 0; v < 8; ++v)
      out[(size_t)(row0 + v + rbase) * (NA * HS) + a * HS + nt * 16 + col] = acc[v] + bias;
  }
}

extern "C" void kernel_launch(void* const* d_in, const int* in_sizes, int n_in,
                              void* d_out, int out_size, void* d_ws, size_t ws_size,
                              hipStream_t stream) {
  const float* nh = (const float*)d_in[0];
  const float* te = (const float*)d_in[1];
  const float* U  = (const float*)d_in[2];
  const float* bU = (const float*)d_in[3];
  const float* Ut = (const float*)d_in[4];
  const float* bt = (const float*)d_in[5];
  const float* T  = (const float*)d_in[6];
  const float* Uo = (const float*)d_in[7];
  const float* bo = (const float*)d_in[8];
  float* out = (float*)d_out;
  _Float16* ws = (_Float16*)d_ws;   // NA * WS_PER_A halfs = 479 KB

  // 1) build pre-swizzled f16 weight image (tiny; 4 blocks)
  hosvd_prep_ws<<<dim3(NA), dim3(256), 0, stream>>>(U, T, Uo, ws);

  // 2) main fused kernel
  const size_t smem =
      (size_t)WS_PER_A * sizeof(_Float16) +                                            // 119808 B
      (size_t)(4 * 16 * 12 + 4 * 16 * 20 + 4 * 17 * 16 + 4 * 16 * 32) * sizeof(float); // 20736 B
  dim3 grid(BS / 64, NA, 1);
  dim3 block(128, 1, 1);   // 4 waves of 32; each wave owns a 16-row batch tile
  hosvd_tucker_wmma<<<grid, block, smem, stream>>>(nh, te, bU, Ut, bt, bo, ws, out);
}
